// RouterKT_80410377715772
// MI455X (gfx1250) — matrix operand order repaired and checked by reading, not compiled, not run on hardware
//
#include <hip/hip_runtime.h>
#include <math.h>
#include <stdint.h>

typedef __attribute__((ext_vector_type(16))) _Float16 v16h;
typedef __attribute__((ext_vector_type(8)))  _Float16 v8h;
typedef __attribute__((ext_vector_type(8)))  float    v8f;
typedef __attribute__((ext_vector_type(4)))  unsigned int u32x4;
typedef __attribute__((ext_vector_type(8)))  int i32x8;
typedef __attribute__((ext_vector_type(4)))  int i32x4;

union V16 { v16h v; v8h h[2]; };

constexpr int B_   = 24;
constexpr int S_   = 512;
constexpr int D_   = 512;
constexpr int H_   = 8;
constexpr int DK_  = 64;
constexpr int DFF_ = 2048;
constexpr int NL_  = 6;
constexpr int FC1_ = 512;
constexpr int FC2_ = 256;
constexpr int BS_  = B_ * S_;
constexpr int HR_  = H_ - 1;   // 7 routed heads
constexpr int BROW_ = 72;      // padded halves per staged B row (144B stride)

#if defined(__HIP_DEVICE_COMPILE__) && defined(__gfx1250__)
#  if __has_builtin(__builtin_amdgcn_tensor_load_to_lds) && \
      __has_builtin(__builtin_amdgcn_s_wait_tensorcnt)
#    define USE_TDM 1
#  endif
#endif
#ifndef USE_TDM
#define USE_TDM 0
#endif

__device__ __forceinline__ v8f wmma_f16(v16h a, v16h b, v8f c) {
  return __builtin_amdgcn_wmma_f32_16x16x32_f16(false, a, false, b, (short)0, c,
                                                false, false);
}

#if USE_TDM
// Issue TDM load of a 32-row x 64-half tile of BT (row stride K halves) into
// LDS at byte offset buf*(32*BROW_*2); rows padded to 144B in LDS.
__device__ __forceinline__ void tdm_issue_b(const _Float16* BT, int cn0, int kc,
                                            int K, int buf) {
  unsigned long long ga =
      (unsigned long long)(uintptr_t)(BT + (size_t)cn0 * K + kc);
  u32x4 g0;
  g0[0] = 1u;                                   // count=1 valid descriptor
  g0[1] = (unsigned)(buf * (32 * BROW_ * 2));   // lds_addr (bytes)
  g0[2] = (unsigned)(ga & 0xFFFFFFFFu);         // global_addr lo
  g0[3] = (unsigned)((ga >> 32) & 0x01FFFFFFu) | 0x80000000u;  // hi | type=2
  i32x8 g1;
  g1[0] = (1 << 16)        // data_size = 2 bytes
        | (1 << 20)        // pad_enable
        | (4 << 22)        // pad_interval: 32 dwords (one 128B row)
        | (3 << 25);       // pad_amount: 4 dwords (16B) -> 144B row stride
  g1[1] = (int)(((unsigned)K & 0xFFFFu) << 16);               // tensor_dim0 lo
  g1[2] = (int)((((unsigned)K >> 16) & 0xFFFFu) | (0xFFFFu << 16)); // d0hi|d1lo
  g1[3] = (int)(64u << 16);     // tensor_dim1 hi=0 | tile_dim0 = 64
  g1[4] = 32;                   // tile_dim1 = 32, tile_dim2 = 0
  g1[5] = K;                    // tensor_dim0_stride lo
  g1[6] = 0;
  g1[7] = 0;
  i32x4 z4 = {0, 0, 0, 0};
#if __clang_major__ >= 23
  i32x8 z8 = {0, 0, 0, 0, 0, 0, 0, 0};
  __builtin_amdgcn_tensor_load_to_lds(g0, g1, z4, z4, z8, 0);
#else
  __builtin_amdgcn_tensor_load_to_lds(g0, g1, z4, z4, 0);
#endif
}
#endif

// ---------------------------------------------------------------------------
// Weight convert fp32 [K][N] -> f16 [N][K] (transposed, for B-fragments)
// ---------------------------------------------------------------------------
__global__ void convert_transpose(const float* __restrict__ src,
                                  _Float16* __restrict__ dst, int K, int N) {
  size_t i = (size_t)blockIdx.x * blockDim.x + threadIdx.x;
  if (i >= (size_t)K * N) return;
  size_t n = i / K, k = i % K;
  dst[i] = (_Float16)src[k * N + n];
}

// fp32 [rows][cols] -> f16 at dst[row*dstStride + dstOff + col]
__global__ void cast_concat(const float* __restrict__ src,
                            _Float16* __restrict__ dst, int rows, int cols,
                            int dstStride, int dstOff) {
  size_t i = (size_t)blockIdx.x * blockDim.x + threadIdx.x;
  if (i >= (size_t)rows * cols) return;
  size_t r = i / cols, c = i % cols;
  dst[r * dstStride + dstOff + c] = (_Float16)src[i];
}

// f16 vl [B,S,H*DK] -> f16 vT [B,H,DK,S]
__global__ void transpose_v(const _Float16* __restrict__ vl,
                            _Float16* __restrict__ vT) {
  size_t i = (size_t)blockIdx.x * blockDim.x + threadIdx.x;
  if (i >= (size_t)BS_ * D_) return;
  int d = (int)(i % D_);
  int t = (int)(i / D_);
  int b = t / S_, s = t % S_;
  int h = d / DK_, dk = d % DK_;
  vT[(((size_t)(b * H_ + h) * DK_) + dk) * S_ + s] = vl[i];
}

// ---------------------------------------------------------------------------
// Rasch embeddings
// ---------------------------------------------------------------------------
__global__ void embed_k(const int* __restrict__ qd, const int* __restrict__ tg,
                        const int* __restrict__ pd,
                        const float* __restrict__ qtab,
                        const float* __restrict__ qatab,
                        const float* __restrict__ qdtab,
                        const float* __restrict__ qadtab,
                        const float* __restrict__ ptab,
                        float* __restrict__ qe_out, float* __restrict__ qa_out) {
  size_t i = (size_t)blockIdx.x * blockDim.x + threadIdx.x;
  if (i >= (size_t)BS_ * D_) return;
  int t = (int)(i / D_), d = (int)(i % D_);
  int qi = qd[t], ti = tg[t], pi = pd[t];
  float qe = qtab[(size_t)qi * D_ + d];
  float qa = qatab[(size_t)ti * D_ + d] + qe;
  float p  = ptab[pi];
  float dq = qdtab[(size_t)qi * D_ + d];
  qe_out[i] = qe + p * dq;
  qa_out[i] = qa + p * (qadtab[(size_t)ti * D_ + d] + dq);
}

// ---------------------------------------------------------------------------
// WMMA GEMM: out[M,N] = A[M,K](f16,row) x BT[N,K](f16,row) + bias, opt relu.
// Block = 8 waves, 256x32 output tile. B tile (32 x 64 halves) staged in LDS
// per workgroup (TDM double-buffered when available), shared by all waves.
// Each wave: 32x32 output = 2x2 register tiles, 4 WMMAs per 32-K step.
// ---------------------------------------------------------------------------
__global__ __launch_bounds__(256) void gemm_wmma_k(
    const _Float16* __restrict__ A, const _Float16* __restrict__ BT,
    const float* __restrict__ bias, float* __restrict__ out32,
    _Float16* __restrict__ out16, int N, int K, int relu) {
  __shared__ _Float16 Bsh[2][32][BROW_];
#if USE_TDM
  // The TDM engine writes Bsh behind the compiler's back (the LDS address is
  // an integer inside the descriptor). Escape the address so the compiler
  // cannot assume Bsh is never written and delete the ds_load reads.
  {
    _Float16* esc = &Bsh[0][0][0];
    asm volatile("" ::"v"(esc) : "memory");
  }
#endif
  const int wv = threadIdx.x >> 5;
  const int lane = threadIdx.x & 31;
  const int lm = lane & 15, g = lane >> 4;
  const int nb = N >> 5;
  const int bm = blockIdx.x / nb, bn = blockIdx.x % nb;
  const int tm0 = bm * 256 + wv * 32;
  const int cn0 = bn * 32;

  v8f acc[2][2];
#pragma unroll
  for (int rt = 0; rt < 2; ++rt)
#pragma unroll
    for (int nt = 0; nt < 2; ++nt)
#pragma unroll
      for (int r = 0; r < 8; ++r) acc[rt][nt][r] = 0.f;

  const _Float16* ap0 = A + (size_t)(tm0 + lm) * K + g * 16;
  const _Float16* ap1 = A + (size_t)(tm0 + 16 + lm) * K + g * 16;

#if USE_TDM
  if (threadIdx.x < 32) tdm_issue_b(BT, cn0, 0, K, 0);
#endif

  for (int kc = 0, it = 0; kc < K; kc += 64, ++it) {
#if USE_TDM
    const int buf = it & 1;
    if (threadIdx.x < 32) __builtin_amdgcn_s_wait_tensorcnt(0);
    __syncthreads();
    if (threadIdx.x < 32 && (kc + 64) < K)
      tdm_issue_b(BT, cn0, kc + 64, K, buf ^ 1);
#else
    const int buf = 0;
    __syncthreads();
    {
      int row = threadIdx.x >> 3, seg = threadIdx.x & 7;
      *(v8h*)&Bsh[0][row][seg * 8] =
          *(const v8h*)(BT + (size_t)(cn0 + row) * K + kc + seg * 8);
    }
    __syncthreads();
#endif
#pragma unroll
    for (int ks = 0; ks < 64; ks += 32) {
      V16 a0, a1, b0, b1;
      a0.h[0] = *(const v8h*)(ap0 + kc + ks);
      a0.h[1] = *(const v8h*)(ap0 + kc + ks + 8);
      a1.h[0] = *(const v8h*)(ap1 + kc + ks);
      a1.h[1] = *(const v8h*)(ap1 + kc + ks + 8);
      const _Float16* bp0 = &Bsh[buf][lm][ks + g * 16];
      const _Float16* bp1 = &Bsh[buf][16 + lm][ks + g * 16];
      b0.h[0] = *(const v8h*)(bp0);
      b0.h[1] = *(const v8h*)(bp0 + 8);
      b1.h[0] = *(const v8h*)(bp1);
      b1.h[1] = *(const v8h*)(bp1 + 8);
      acc[0][0] = wmma_f16(a0.v, b0.v, acc[0][0]);
      acc[0][1] = wmma_f16(a0.v, b1.v, acc[0][1]);
      acc[1][0] = wmma_f16(a1.v, b0.v, acc[1][0]);
      acc[1][1] = wmma_f16(a1.v, b1.v, acc[1][1]);
    }
  }

#pragma unroll
  for (int nt = 0; nt < 2; ++nt) {
    const int col = cn0 + nt * 16 + lm;
    const float bval = bias ? bias[col] : 0.f;
#pragma unroll
    for (int rt = 0; rt < 2; ++rt) {
#pragma unroll
      for (int r = 0; r < 8; ++r) {
        float v = acc[rt][nt][r] + bval;
        if (relu) v = fmaxf(v, 0.f);
        size_t row = (size_t)(tm0 + rt * 16 + r + 8 * g);
        if (out32) out32[row * N + col] = v;
        if (out16) out16[row * N + col] = (_Float16)v;
      }
    }
  }
}

// ---------------------------------------------------------------------------
// Router: gates = softmax(ql @ Wg), top-2 mask -> dyn scores; then per-batch
// mean over sequence -> rmask[B,H] with identity head 0 fixed to 1.
// ---------------------------------------------------------------------------
__global__ __launch_bounds__(64) void router_gates(const float* __restrict__ ql,
                                                   const float* __restrict__ Wg,
                                                   float* __restrict__ dyn) {
  __shared__ float red[64][8];
  const int t = blockIdx.x, tid = threadIdx.x;
  float acc[HR_];
#pragma unroll
  for (int k = 0; k < HR_; ++k) acc[k] = 0.f;
  for (int d = tid; d < D_; d += 64) {
    float xv = ql[(size_t)t * D_ + d];
#pragma unroll
    for (int k = 0; k < HR_; ++k) acc[k] += xv * Wg[(size_t)d * HR_ + k];
  }
#pragma unroll
  for (int k = 0; k < HR_; ++k) red[tid][k] = acc[k];
  __syncthreads();
  if (tid == 0) {
    float gsum[HR_];
#pragma unroll
    for (int k = 0; k < HR_; ++k) {
      float s = 0.f;
      for (int j = 0; j < 64; ++j) s += red[j][k];
      gsum[k] = s;
    }
    float mx = gsum[0];
#pragma unroll
    for (int k = 1; k < HR_; ++k) mx = fmaxf(mx, gsum[k]);
    float e[HR_];
    float den = 0.f;
#pragma unroll
    for (int k = 0; k < HR_; ++k) { e[k] = __expf(gsum[k] - mx); den += e[k]; }
#pragma unroll
    for (int k = 0; k < HR_; ++k) e[k] /= den;
    int i1 = 0;
#pragma unroll
    for (int k = 1; k < HR_; ++k) if (e[k] > e[i1]) i1 = k;
    int i2 = (i1 == 0) ? 1 : 0;
#pragma unroll
    for (int k = 0; k < HR_; ++k)
      if (k != i1 && e[k] > e[i2]) i2 = k;
#pragma unroll
    for (int k = 0; k < HR_; ++k)
      dyn[(size_t)t * HR_ + k] = (k == i1 || k == i2) ? e[k] : 0.f;
  }
}

__global__ __launch_bounds__(256) void router_rmask(const float* __restrict__ dyn,
                                                    float* __restrict__ rmask) {
  __shared__ float red[256][8];
  const int b = blockIdx.x, tid = threadIdx.x;
  float acc[HR_];
#pragma unroll
  for (int k = 0; k < HR_; ++k) acc[k] = 0.f;
  for (int s = tid; s < S_; s += 256) {
#pragma unroll
    for (int k = 0; k < HR_; ++k)
      acc[k] += dyn[((size_t)(b * S_ + s)) * HR_ + k];
  }
#pragma unroll
  for (int k = 0; k < HR_; ++k) red[tid][k] = acc[k];
  __syncthreads();
  for (int st = 128; st > 0; st >>= 1) {
    if (tid < st) {
#pragma unroll
      for (int k = 0; k < HR_; ++k) red[tid][k] += red[tid + st][k];
    }
    __syncthreads();
  }
  if (tid == 0) {
    rmask[b * H_] = 1.f;
#pragma unroll
    for (int k = 0; k < HR_; ++k)
      rmask[b * H_ + 1 + k] = red[0][k] * (1.f / (float)S_);
  }
}

// ---------------------------------------------------------------------------
// Flash attention with WMMA. One wave per (b,h, 16-row tile). k == q.
// scores = q qT / 8, causal (strict or incl-self), row 0 -> all-zero scores.
// Output scaled by rmask[b,h], written f16.
// ---------------------------------------------------------------------------
__global__ __launch_bounds__(128) void attn_wmma(
    const _Float16* __restrict__ q16, const _Float16* __restrict__ vT16,
    const float* __restrict__ rmask, _Float16* __restrict__ ctx16, int strict) {
  __shared__ _Float16 Psh[4][16][48];
  const int w = threadIdx.x >> 5;
  const int lane = threadIdx.x & 31;
  const int lm = lane & 15, g = lane >> 4;
  const int tile = blockIdx.x * 4 + w;
  const int i0 = (tile & 31) * 16;        // S_/16 == 32 row tiles
  const int bh = tile >> 5;
  const int b = bh / H_, h = bh % H_;

  // Q A-fragments for rows i0..i0+15, K chunks 0-31 and 32-63
  const _Float16* qa = q16 + ((size_t)(b * S_ + i0 + lm)) * D_ + h * DK_ + g * 16;
  V16 aq[2];
  aq[0].h[0] = *(const v8h*)(qa);
  aq[0].h[1] = *(const v8h*)(qa + 8);
  aq[1].h[0] = *(const v8h*)(qa + 32);
  aq[1].h[1] = *(const v8h*)(qa + 40);

  v8f O[4];
  float m[8], l[8];
#pragma unroll
  for (int r = 0; r < 8; ++r) { m[r] = -1e30f; l[r] = 0.f; }
#pragma unroll
  for (int nt = 0; nt < 4; ++nt)
#pragma unroll
    for (int r = 0; r < 8; ++r) O[nt][r] = 0.f;

  for (int jc = 0; jc < S_; jc += 32) {
#pragma unroll
    for (int sub = 0; sub < 32; sub += 16) {
      const int j0 = jc + sub;
      const _Float16* qb =
          q16 + ((size_t)(b * S_ + j0 + lm)) * D_ + h * DK_ + g * 16;
      V16 b0, b1;
      b0.h[0] = *(const v8h*)(qb);
      b0.h[1] = *(const v8h*)(qb + 8);
      b1.h[0] = *(const v8h*)(qb + 32);
      b1.h[1] = *(const v8h*)(qb + 40);
      v8f s;
#pragma unroll
      for (int r = 0; r < 8; ++r) s[r] = 0.f;
      s = wmma_f16(aq[0].v, b0.v, s);
      s = wmma_f16(aq[1].v, b1.v, s);

      const int j = j0 + lm;
#pragma unroll
      for (int r = 0; r < 8; ++r) {
        const int i = i0 + r + 8 * g;
        float sv = s[r] * 0.125f;
        bool allowed = strict ? (j < i) : (j <= i);
        sv = allowed ? sv : -1e9f;
        sv = (i == 0) ? 0.f : sv;     // pad_zero row -> uniform softmax
        float rm_ = sv;
        rm_ = fmaxf(rm_, __shfl_xor(rm_, 1));
        rm_ = fmaxf(rm_, __shfl_xor(rm_, 2));
        rm_ = fmaxf(rm_, __shfl_xor(rm_, 4));
        rm_ = fmaxf(rm_, __shfl_xor(rm_, 8));
        float nm = fmaxf(m[r], rm_);
        float p = __expf(sv - nm);
        float rs = p;
        rs += __shfl_xor(rs, 1);
        rs += __shfl_xor(rs, 2);
        rs += __shfl_xor(rs, 4);
        rs += __shfl_xor(rs, 8);
        float al = __expf(m[r] - nm);
        l[r] = l[r] * al + rs;
        m[r] = nm;
#pragma unroll
        for (int nt = 0; nt < 4; ++nt) O[nt][r] *= al;
        Psh[w][r + 8 * g][sub + lm] = (_Float16)p;
      }
    }
    // P (16x32, f16) as A-fragment from LDS; PV over K=32
    V16 pa;
    pa.h[0] = *(const v8h*)&Psh[w][lm][g * 16];
    pa.h[1] = *(const v8h*)&Psh[w][lm][g * 16 + 8];
#pragma unroll
    for (int nt = 0; nt < 4; ++nt) {
      const _Float16* vb =
          vT16 + (((size_t)(b * H_ + h) * DK_) + nt * 16 + lm) * S_ + jc + g * 16;
      V16 bv;
      bv.h[0] = *(const v8h*)(vb);
      bv.h[1] = *(const v8h*)(vb + 8);
      O[nt] = wmma_f16(pa.v, bv.v, O[nt]);
    }
  }

  const float rmv = rmask[b * H_ + h];
#pragma unroll
  for (int r = 0; r < 8; ++r) {
    const float inv = rmv / l[r];
    const size_t row = (size_t)(b * S_ + i0 + r + 8 * g);
#pragma unroll
    for (int nt = 0; nt < 4; ++nt)
      ctx16[row * D_ + h * DK_ + nt * 16 + lm] = (_Float16)(O[nt][r] * inv);
  }
}

// ---------------------------------------------------------------------------
// LayerNorm(x + res) * g + b ; one block per token, D=512, 256 threads
// ---------------------------------------------------------------------------
__global__ __launch_bounds__(256) void ln_residual(
    const float* __restrict__ x, const float* __restrict__ res,
    const float* __restrict__ gam, const float* __restrict__ bet,
    float* __restrict__ out32, _Float16* __restrict__ out16) {
  __shared__ float sA[256], sB[256];
  const int t = blockIdx.x, tid = threadIdx.x;
  const size_t base = (size_t)t * D_;
  float a0 = x[base + tid] + res[base + tid];
  float a1 = x[base + tid + 256] + res[base + tid + 256];
  sA[tid] = a0 + a1;
  sB[tid] = a0 * a0 + a1 * a1;
  __syncthreads();
  for (int st = 128; st > 0; st >>= 1) {
    if (tid < st) { sA[tid] += sA[tid + st]; sB[tid] += sB[tid + st]; }
    __syncthreads();
  }
  const float mean = sA[0] * (1.f / (float)D_);
  const float var = sB[0] * (1.f / (float)D_) - mean * mean;
  const float rstd = rsqrtf(var + 1e-5f);
  const float o0 = (a0 - mean) * rstd * gam[tid] + bet[tid];
  const float o1 = (a1 - mean) * rstd * gam[tid + 256] + bet[tid + 256];
  if (out32) { out32[base + tid] = o0; out32[base + tid + 256] = o1; }
  if (out16) {
    out16[base + tid] = (_Float16)o0;
    out16[base + tid + 256] = (_Float16)o1;
  }
}

// ---------------------------------------------------------------------------
// Final head: sigmoid(h2 @ ow3 + ob3), one thread per token
// ---------------------------------------------------------------------------
__global__ void head_final(const _Float16* __restrict__ h2,
                           const float* __restrict__ w3,
                           const float* __restrict__ b3,
                           float* __restrict__ out, int n) {
  const int t = blockIdx.x * blockDim.x + threadIdx.x;
  if (t >= n) return;
  float acc = b3[0];
  for (int c = 0; c < FC2_; ++c)
    acc += (float)h2[(size_t)t * FC2_ + c] * w3[c];
  out[t] = 1.f / (1.f + __expf(-acc));
}

// ---------------------------------------------------------------------------
// Host orchestration
// ---------------------------------------------------------------------------
extern "C" void kernel_launch(void* const* d_in, const int* in_sizes, int n_in,
                              void* d_out, int out_size, void* d_ws,
                              size_t ws_size, hipStream_t stream) {
  (void)in_sizes; (void)n_in; (void)out_size; (void)ws_size;

  const int* q_data   = (const int*)d_in[0];
  const int* target   = (const int*)d_in[1];
  const int* pid_data = (const int*)d_in[2];
  const float* q_tab  = (const float*)d_in[3];
  const float* qa_tab = (const float*)d_in[4];
  const float* qd_tab = (const float*)d_in[5];
  const float* qad_tab= (const float*)d_in[6];
  const float* pid_tab= (const float*)d_in[7];
  const float* Wq = (const float*)d_in[8];
  const float* bq = (const float*)d_in[9];
  const float* Wv = (const float*)d_in[10];
  const float* bv = (const float*)d_in[11];
  const float* Wg = (const float*)d_in[12];
  const float* Wo = (const float*)d_in[13];
  const float* bo = (const float*)d_in[14];
  const float* ln1_g = (const float*)d_in[15];
  const float* ln1_b = (const float*)d_in[16];
  const float* W1 = (const float*)d_in[17];
  const float* b1 = (const float*)d_in[18];
  const float* W2 = (const float*)d_in[19];
  const float* b2 = (const float*)d_in[20];
  const float* ln2_g = (const float*)d_in[21];
  const float* ln2_b = (const float*)d_in[22];
  const float* ow1 = (const float*)d_in[23];
  const float* ob1 = (const float*)d_in[24];
  const float* ow2 = (const float*)d_in[25];
  const float* ob2 = (const float*)d_in[26];
  const float* ow3 = (const float*)d_in[27];
  const float* ob3 = (const float*)d_in[28];

  char* wsp = (char*)d_ws;
  size_t off = 0;
  auto alloc = [&](size_t bytes) -> void* {
    void* p = wsp + off;
    off = (off + bytes + 255) & ~(size_t)255;
    return p;
  };

  // f16 transposed weights
  _Float16* WqT  = (_Float16*)alloc((size_t)NL_ * D_ * D_ * 2);
  _Float16* WvT  = (_Float16*)alloc((size_t)NL_ * D_ * D_ * 2);
  _Float16* WoT  = (_Float16*)alloc((size_t)NL_ * D_ * D_ * 2);
  _Float16* W1T  = (_Float16*)alloc((size_t)NL_ * D_ * DFF_ * 2);
  _Float16* W2T  = (_Float16*)alloc((size_t)NL_ * DFF_ * D_ * 2);
  _Float16* ow1T = (_Float16*)alloc((size_t)2 * D_ * FC1_ * 2);
  _Float16* ow2T = (_Float16*)alloc((size_t)FC1_ * FC2_ * 2);

  // fp32 activations
  float* q_emb32 = (float*)alloc((size_t)BS_ * D_ * 4);
  float* qa_emb32= (float*)alloc((size_t)BS_ * D_ * 4);
  float* y32     = (float*)alloc((size_t)BS_ * D_ * 4);
  float* x32     = (float*)alloc((size_t)BS_ * D_ * 4);
  float* ql32    = (float*)alloc((size_t)BS_ * D_ * 4);   // also o32/f2_32
  float* x1_32   = (float*)alloc((size_t)BS_ * D_ * 4);

  // f16 activations
  _Float16* a16   = (_Float16*)alloc((size_t)BS_ * D_ * 2);
  _Float16* b16   = (_Float16*)alloc((size_t)BS_ * D_ * 2);
  _Float16* ql16  = (_Float16*)alloc((size_t)BS_ * D_ * 2);
  _Float16* vl16  = (_Float16*)alloc((size_t)BS_ * D_ * 2);
  _Float16* vT16  = (_Float16*)alloc((size_t)BS_ * D_ * 2);
  _Float16* ctx16 = (_Float16*)alloc((size_t)BS_ * D_ * 2);
  _Float16* x1_16 = (_Float16*)alloc((size_t)BS_ * D_ * 2);
  _Float16* h16ff = (_Float16*)alloc((size_t)BS_ * DFF_ * 2); // also cat16

  float* dyn      = (float*)alloc((size_t)BS_ * HR_ * 4);
  float* rmaskbuf = (float*)alloc((size_t)B_ * H_ * 4);

  float* o32   = ql32;   // aliases (sequentially dead)
  float* f2_32 = ql32;
  _Float16* cat16 = h16ff;
  _Float16* h1_16 = a16;
  _Float16* h2_16 = ql16;

  auto cvt = [&](const float* src, _Float16* dst, int K, int N) {
    size_t n = (size_t)K * N;
    convert_transpose<<<(unsigned)((n + 255) / 256), 256, 0, stream>>>(src, dst, K, N);
  };
  for (int l = 0; l < NL_; ++l) {
    cvt(Wq + (size_t)l * D_ * D_,   WqT + (size_t)l * D_ * D_,   D_, D_);
    cvt(Wv + (size_t)l * D_ * D_,   WvT + (size_t)l * D_ * D_,   D_, D_);
    cvt(Wo + (size_t)l * D_ * D_,   WoT + (size_t)l * D_ * D_,   D_, D_);
    cvt(W1 + (size_t)l * D_ * DFF_, W1T + (size_t)l * D_ * DFF_, D_, DFF_);
    cvt(W2 + (size_t)l * DFF_ * D_, W2T + (size_t)l * DFF_ * D_, DFF_, D_);
  }
  cvt(ow1, ow1T, 2 * D_, FC1_);
  cvt(ow2, ow2T, FC1_, FC2_);

  // embeddings
  {
    size_t n = (size_t)BS_ * D_;
    embed_k<<<(unsigned)((n + 255) / 256), 256, 0, stream>>>(
        q_data, target, pid_data, q_tab, qa_tab, qd_tab, qad_tab, pid_tab,
        q_emb32, qa_emb32);
  }

  auto gemm = [&](const _Float16* A, const _Float16* BT, const float* bias,
                  float* o32p, _Float16* o16p, int N, int K, int relu) {
    int blocks = (BS_ / 256) * (N / 32);
    gemm_wmma_k<<<blocks, 256, 0, stream>>>(A, BT, bias, o32p, o16p, N, K, relu);
  };
  auto cast = [&](const float* src, _Float16* dst, int rows, int cols,
                  int dstStride, int dstOff) {
    size_t n = (size_t)rows * cols;
    cast_concat<<<(unsigned)((n + 255) / 256), 256, 0, stream>>>(
        src, dst, rows, cols, dstStride, dstOff);
  };

  const float* ycur = qa_emb32;
  const float* xcur = q_emb32;
  for (int i = 0; i < NL_; ++i) {
    const float *xq, *xv;
    int strict;
    float* outb;
    if (i < 2)               { xq = ycur; xv = ycur; strict = 0; outb = y32; }
    else if (i == 2 || i == 4){ xq = xcur; xv = xcur; strict = 0; outb = x32; }
    else                     { xq = xcur; xv = ycur; strict = 1; outb = x32; }

    cast(xq, a16, BS_, D_, D_, 0);
    const _Float16* xv16 = a16;
    if (xv != xq) { cast(xv, b16, BS_, D_, D_, 0); xv16 = b16; }

    gemm(a16,  WqT + (size_t)i * D_ * D_, bq + (size_t)i * D_, ql32, ql16, D_, D_, 0);
    gemm(xv16, WvT + (size_t)i * D_ * D_, bv + (size_t)i * D_, nullptr, vl16, D_, D_, 0);
    {
      size_t n = (size_t)BS_ * D_;
      transpose_v<<<(unsigned)((n + 255) / 256), 256, 0, stream>>>(vl16, vT16);
    }
    router_gates<<<BS_, 64, 0, stream>>>(ql32, Wg + (size_t)i * D_ * HR_, dyn);
    router_rmask<<<B_, 256, 0, stream>>>(dyn, rmaskbuf);
    attn_wmma<<<(B_ * H_ * (S_ / 16)) / 4, 128, 0, stream>>>(
        ql16, vT16, rmaskbuf, ctx16, strict);
    gemm(ctx16, WoT + (size_t)i * D_ * D_, bo + (size_t)i * D_, o32, nullptr, D_, D_, 0);
    ln_residual<<<BS_, 256, 0, stream>>>(o32, xq, ln1_g + (size_t)i * D_,
                                         ln1_b + (size_t)i * D_, x1_32, x1_16);
    gemm(x1_16, W1T + (size_t)i * D_ * DFF_, b1 + (size_t)i * DFF_, nullptr,
         h16ff, DFF_, D_, 1);
    gemm(h16ff, W2T + (size_t)i * DFF_ * D_, b2 + (size_t)i * D_, f2_32,
         nullptr, D_, DFF_, 0);
    ln_residual<<<BS_, 256, 0, stream>>>(f2_32, x1_32, ln2_g + (size_t)i * D_,
                                         ln2_b + (size_t)i * D_, outb, nullptr);
    if (i < 2) ycur = y32; else xcur = x32;
  }

  // output head: concat [x, q_emb] -> FC1 -> FC2 -> 1 -> sigmoid
  cast(xcur,    cat16, BS_, D_, 2 * D_, 0);
  cast(q_emb32, cat16, BS_, D_, 2 * D_, D_);
  gemm(cat16, ow1T, ob1, nullptr, h1_16, FC1_, 2 * D_, 1);
  gemm(h1_16, ow2T, ob2, nullptr, h2_16, FC2_, FC1_, 1);
  head_final<<<(BS_ + 255) / 256, 256, 0, stream>>>(h2_16, ow3, ob3,
                                                    (float*)d_out, BS_);
}
